// GATDecoder_4492535791746
// MI455X (gfx1250) — compile-verified, matrix operand order checked by reading, not run on hardware
//
#include <hip/hip_runtime.h>
#include <hip/hip_bf16.h>
#include <math.h>

typedef __attribute__((ext_vector_type(16))) _Float16 v16h;
typedef __attribute__((ext_vector_type(8)))  float    v8f;

#define NN    50000
#define EE    600000
#define E2    (EE + NN)
#define DIMS  128
#define BB    1024
#define NEG_GAT 0.2f
#define NEG_DEC 0.1f

// ---------- helpers ----------

__device__ inline v16h load16h(const float* __restrict__ p, int kb) {
  // loads p[kb..kb+7] and p[kb+16..kb+23] as halves (ISA 16-bit A/B layout)
  const float4* q0 = (const float4*)(p + kb);
  const float4* q1 = (const float4*)(p + kb + 16);
  float4 a0 = q0[0], a1 = q0[1], b0 = q1[0], b1 = q1[1];
  v16h r;
  r[0]=(_Float16)a0.x; r[1]=(_Float16)a0.y; r[2]=(_Float16)a0.z; r[3]=(_Float16)a0.w;
  r[4]=(_Float16)a1.x; r[5]=(_Float16)a1.y; r[6]=(_Float16)a1.z; r[7]=(_Float16)a1.w;
  r[8]=(_Float16)b0.x; r[9]=(_Float16)b0.y; r[10]=(_Float16)b0.z; r[11]=(_Float16)b0.w;
  r[12]=(_Float16)b1.x; r[13]=(_Float16)b1.y; r[14]=(_Float16)b1.z; r[15]=(_Float16)b1.w;
  return r;
}

// hardware f32 atomic add (global_atomic_add_f32), avoid CAS-loop fallback
__device__ inline void atomicAddF(float* addr, float val) {
  unsafeAtomicAdd(addr, val);
}

__device__ inline void atomicMaxF(float* addr, float val) {
  if (val >= 0.0f) atomicMax((int*)addr, __float_as_int(val));
  else             atomicMin((unsigned int*)addr, __float_as_uint(val));
}

__device__ inline void edge_sd(const int* __restrict__ EI, int e, int& s, int& d) {
  if (e < EE) { s = EI[e]; d = EI[EE + e]; }
  else        { s = e - EE; d = e - EE; }   // self loops
}

// ---------- kernels ----------

// H[m][n] = sum_k X[m][k] * W[n][k]   (X: nrows x 128, W: 128 x 128, both row-major f32)
__global__ void gat_gemm_xwt(const float* __restrict__ X, const float* __restrict__ W,
                             float* __restrict__ H, int nrows) {
  int wave = (blockIdx.x * blockDim.x + threadIdx.x) >> 5;
  int lane = threadIdx.x & 31;
  if (wave * 16 >= nrows) return;          // wave-uniform exit, EXEC stays all-ones
  int half = lane >> 4;
  int r    = lane & 15;
  const float* arow = X + (size_t)(wave * 16 + r) * DIMS;
  int kb0 = half * 8;

  // preload + convert all 4 K-step A fragments once (reused by all 8 N-tiles)
  v16h a[4];
  #pragma unroll
  for (int kk = 0; kk < 4; ++kk) a[kk] = load16h(arow, kk * 32 + kb0);

  #pragma unroll
  for (int nt = 0; nt < 8; ++nt) {
    const float* brow = W + (size_t)(nt * 16 + r) * DIMS;
    v8f c = {};
    #pragma unroll
    for (int kk = 0; kk < 4; ++kk) {
      v16h b = load16h(brow, kk * 32 + kb0);
      c = __builtin_amdgcn_wmma_f32_16x16x32_f16(false, a[kk], false, b, (short)0, c,
                                                 false, false);
    }
    #pragma unroll
    for (int rr = 0; rr < 8; ++rr) {
      int row = wave * 16 + rr + half * 8;
      H[(size_t)row * DIMS + nt * 16 + r] = c[rr];
    }
  }
}

// s1[n] = dot(H[n], att[0:128]) (dst side);  s2[n] = dot(H[n], att[128:256]) (src side)
__global__ void gat_node_scores(const float* __restrict__ H, const float* __restrict__ att,
                                float* __restrict__ s1, float* __restrict__ s2) {
  int node = blockIdx.x, t = threadIdx.x;
  float v = H[(size_t)node * DIMS + t];
  __shared__ float r1[DIMS], r2[DIMS];
  r1[t] = v * att[t];
  r2[t] = v * att[DIMS + t];
  __syncthreads();
  for (int s = 64; s > 0; s >>= 1) {
    if (t < s) { r1[t] += r1[t + s]; r2[t] += r2[t + s]; }
    __syncthreads();
  }
  if (t == 0) { s1[node] = r1[0]; s2[node] = r2[0]; }
}

// zero agg / den / accum, set amax = -inf
__global__ void gat_layer_init(float* __restrict__ agg, float* __restrict__ amax,
                               float* __restrict__ den, float* __restrict__ accum) {
  int idx = blockIdx.x * blockDim.x + threadIdx.x;
  if (idx < NN * DIMS) agg[idx] = 0.0f;
  if (idx < NN) { amax[idx] = -__builtin_inff(); den[idx] = 0.0f; }
  if (idx < 2 * DIMS) accum[idx] = 0.0f;
}

__global__ void gat_edge_alpha_max(const int* __restrict__ EI, const float* __restrict__ s1,
                                   const float* __restrict__ s2, float* __restrict__ amax) {
  int e = blockIdx.x * blockDim.x + threadIdx.x;
  if (e >= E2) return;
  int s, d; edge_sd(EI, e, s, d);
  float a = s1[d] + s2[s];
  a = (a > 0.0f) ? a : NEG_GAT * a;
  atomicMaxF(&amax[s], a);
}

__global__ void gat_edge_exp(const int* __restrict__ EI, const float* __restrict__ s1,
                             const float* __restrict__ s2, const float* __restrict__ amax,
                             float* __restrict__ den, float* __restrict__ exbuf) {
  int e = blockIdx.x * blockDim.x + threadIdx.x;
  if (e >= E2) return;
  int s, d; edge_sd(EI, e, s, d);
  float a = s1[d] + s2[s];
  a = (a > 0.0f) ? a : NEG_GAT * a;
  float m = amax[s];
  if (!(m > -1e37f && m < 1e37f)) m = 0.0f;     // where(isfinite(amax), amax, 0)
  float ex = __expf(a - m);
  exbuf[e] = ex;
  atomicAddF(&den[s], ex);
}

// agg[dst] += h[src] * (ex / (den[src] + 1e-16))
__global__ void gat_edge_scatter(const float* __restrict__ H, const int* __restrict__ EI,
                                 const float* __restrict__ exbuf, const float* __restrict__ den,
                                 float* __restrict__ agg) {
  long long tid = (long long)blockIdx.x * blockDim.x + threadIdx.x;
  int e = (int)(tid >> 7);
  if (e >= E2) return;
  int f = (int)(tid & 127);
  int s, d; edge_sd(EI, e, s, d);
  float w = exbuf[e] / (den[s] + 1e-16f);
  atomicAddF(&agg[(size_t)d * DIMS + f], H[(size_t)s * DIMS + f] * w);
}

// per-feature sum / sumsq of v = relu(agg + bias)
#define BN_ROWS 256
__global__ void gat_bn_stats(const float* __restrict__ agg, const float* __restrict__ bias,
                             float* __restrict__ accum) {
  int f = threadIdx.x;                       // 128 threads = features
  int r0 = blockIdx.x * BN_ROWS;
  int r1 = r0 + BN_ROWS; if (r1 > NN) r1 = NN;
  float bf = bias[f];
  float s = 0.0f, q = 0.0f;
  for (int r = r0; r < r1; ++r) {
    float v = agg[(size_t)r * DIMS + f] + bf;
    v = (v > 0.0f) ? v : 0.0f;
    s += v; q += v * v;
  }
  atomicAddF(&accum[f], s);
  atomicAddF(&accum[DIMS + f], q);
}

// in-place: agg <- leaky_relu( bn(relu(agg+bias)), 0.1 )
__global__ void gat_bn_norm(float* __restrict__ agg, const float* __restrict__ bias,
                            const float* __restrict__ accum, const float* __restrict__ g,
                            const float* __restrict__ be) {
  int idx = blockIdx.x * blockDim.x + threadIdx.x;
  if (idx >= NN * DIMS) return;
  int f = idx & 127;
  float v = agg[idx] + bias[f];
  v = (v > 0.0f) ? v : 0.0f;
  const float invn = 1.0f / (float)NN;
  float mu  = accum[f] * invn;
  float var = accum[DIMS + f] * invn - mu * mu;
  float y = (v - mu) * rsqrtf(var + 1e-5f) * g[f] + be[f];
  agg[idx] = (y > 0.0f) ? y : NEG_DEC * y;
}

// M = P1 @ P2 @ P1^T   (P1: 128x64, P2: 64x64, M: 128x128)
__global__ void dec_build_M(const float* __restrict__ P1, const float* __restrict__ P2,
                            float* __restrict__ M) {
  __shared__ float T[DIMS * 64];             // P1 @ P2, 32 KB
  int tid = threadIdx.x;                     // 256 threads
  for (int idx = tid; idx < DIMS * 64; idx += 256) {
    int i = idx >> 6, j = idx & 63;
    float acc = 0.0f;
    #pragma unroll 8
    for (int k = 0; k < 64; ++k) acc += P1[i * 64 + k] * P2[k * 64 + j];
    T[idx] = acc;
  }
  __syncthreads();
  for (int idx = tid; idx < DIMS * DIMS; idx += 256) {
    int i = idx >> 7, j = idx & 127;
    float acc = 0.0f;
    #pragma unroll 8
    for (int k = 0; k < 64; ++k) acc += T[i * 64 + k] * P1[j * 64 + k];
    M[idx] = acc;
  }
}

// ypred[b] = a^T M b   with a = h[di0-1], b = h[di1-1]
__global__ void dec_pairs(const float* __restrict__ H, const float* __restrict__ M,
                          const int* __restrict__ DI, float* __restrict__ out) {
  int b = blockIdx.x, t = threadIdx.x;       // 128 threads
  int na = DI[b * 2 + 0] - 1;
  int nb = DI[b * 2 + 1] - 1;
  __shared__ float bv[DIMS];
  __shared__ float red[DIMS];
  bv[t] = H[(size_t)nb * DIMS + t];
  __syncthreads();
  float rowdot = 0.0f;
  #pragma unroll 4
  for (int j = 0; j < DIMS; ++j) rowdot += M[t * DIMS + j] * bv[j];
  red[t] = H[(size_t)na * DIMS + t] * rowdot;
  __syncthreads();
  for (int s = 64; s > 0; s >>= 1) {
    if (t < s) red[t] += red[t + s];
    __syncthreads();
  }
  if (t == 0) out[b] = red[0];
}

// ---------- launch ----------

extern "C" void kernel_launch(void* const* d_in, const int* in_sizes, int n_in,
                              void* d_out, int out_size, void* d_ws, size_t ws_size,
                              hipStream_t stream) {
  const float* x   = (const float*)d_in[0];
  const int*   EI  = (const int*)d_in[1];
  const int*   DI  = (const int*)d_in[2];
  const float* Wm[3]  = { (const float*)d_in[3],  (const float*)d_in[8],  (const float*)d_in[13] };
  const float* att[3] = { (const float*)d_in[4],  (const float*)d_in[9],  (const float*)d_in[14] };
  const float* bia[3] = { (const float*)d_in[5],  (const float*)d_in[10], (const float*)d_in[15] };
  const float* gam[3] = { (const float*)d_in[6],  (const float*)d_in[11], (const float*)d_in[16] };
  const float* bet[3] = { (const float*)d_in[7],  (const float*)d_in[12], (const float*)d_in[17] };
  const float* P1 = (const float*)d_in[18];
  const float* P2 = (const float*)d_in[19];
  float* out = (float*)d_out;

  float* ws    = (float*)d_ws;
  float* buf0  = ws;                         // N*128  GEMM output h
  float* buf1  = buf0 + (size_t)NN * DIMS;   // N*128  agg / layer output
  float* s1    = buf1 + (size_t)NN * DIMS;   // N
  float* s2    = s1 + NN;                    // N
  float* amax  = s2 + NN;                    // N
  float* den   = amax + NN;                  // N
  float* exbuf = den + NN;                   // E2
  float* accum = exbuf + E2;                 // 256
  float* Mmat  = accum + 2 * DIMS;           // 128*128

  const int tiles = NN / 16;                 // 3125
  const int gemm_blocks = (tiles + 3) / 4;   // 4 waves/block
  const int elem_blocks = (NN * DIMS + 255) / 256;
  const int edge_blocks = (E2 + 255) / 256;
  const long long scat_threads = (long long)E2 * DIMS;
  const int scat_blocks = (int)((scat_threads + 255) / 256);
  const int stat_blocks = (NN + BN_ROWS - 1) / BN_ROWS;

  const float* hin = x;
  for (int l = 0; l < 3; ++l) {
    gat_gemm_xwt<<<gemm_blocks, 128, 0, stream>>>(hin, Wm[l], buf0, NN);
    gat_layer_init<<<elem_blocks, 256, 0, stream>>>(buf1, amax, den, accum);
    gat_node_scores<<<NN, DIMS, 0, stream>>>(buf0, att[l], s1, s2);
    gat_edge_alpha_max<<<edge_blocks, 256, 0, stream>>>(EI, s1, s2, amax);
    gat_edge_exp<<<edge_blocks, 256, 0, stream>>>(EI, s1, s2, amax, den, exbuf);
    gat_edge_scatter<<<scat_blocks, 256, 0, stream>>>(buf0, EI, exbuf, den, buf1);
    gat_bn_stats<<<stat_blocks, DIMS, 0, stream>>>(buf1, bia[l], accum);
    gat_bn_norm<<<elem_blocks, 256, 0, stream>>>(buf1, bia[l], accum, gam[l], bet[l]);
    hin = buf1;
  }

  dec_build_M<<<1, 256, 0, stream>>>(P1, P2, Mmat);
  dec_pairs<<<BB, DIMS, 0, stream>>>(buf1, Mmat, DI, out);
}
